// TorchModel_89137751261740
// MI455X (gfx1250) — matrix operand order repaired and checked by reading, not compiled, not run on hardware
//
#include <hip/hip_runtime.h>
#include <hip/hip_bf16.h>
#include <math.h>

// ---------------------------------------------------------------------------
// Types
// ---------------------------------------------------------------------------
typedef __attribute__((ext_vector_type(16))) __bf16        v16bf;
typedef __attribute__((ext_vector_type(8)))  __bf16        v8bf;
typedef __attribute__((ext_vector_type(8)))  float         v8f;
typedef __attribute__((ext_vector_type(4)))  unsigned int  v4u;
typedef __attribute__((ext_vector_type(8)))  int           v8i;
typedef __attribute__((ext_vector_type(4)))  int           v4i;
typedef __attribute__((ext_vector_type(4)))  unsigned int  u32x4;

struct U32x8 { u32x4 a, b; };   // 32 bytes -> bit_cast to v16bf

#define DHID   256
#define SEQLEN 128
#define NB     16                 // batch rows per workgroup
#define HROW   264                // padded row stride in ushorts (528 B = 512 + 16)
#define TILE_BYTES (DHID * NB * 2) // 8192 B: one bf16 x_proj tile, dense [n][j]
#define XP_BYTES (4096ull * SEQLEN * DHID * 2)  // 256 MiB x_proj workspace
#define VOCAB  50000

// ---------------------------------------------------------------------------
// TDM issue helper (arity differs between ROCm7.2/clang-22 and clang-23)
// ---------------------------------------------------------------------------
static __device__ __forceinline__ void tdm_issue(v4u g0, v8i g1, v4i g2, v4i g3,
                                                 bool store) {
#if defined(__clang_major__) && (__clang_major__ >= 23)
  v8i g4 = (v8i)0;
  if (store) __builtin_amdgcn_tensor_store_from_lds(g0, g1, g2, g3, g4, 0);
  else       __builtin_amdgcn_tensor_load_to_lds  (g0, g1, g2, g3, g4, 0);
#else
  if (store) __builtin_amdgcn_tensor_store_from_lds(g0, g1, g2, g3, 0);
  else       __builtin_amdgcn_tensor_load_to_lds  (g0, g1, g2, g3, 0);
#endif
}

// 1-D dense-global <-> padded-LDS row copy: 16 rows x 512 B, LDS stride 528 B,
// via TDM iterate mode (load or store).
static __device__ __forceinline__ void tdm_rows16(unsigned lds,
                                                  unsigned long long ga,
                                                  bool store) {
  v4u g0 = { 1u, lds, (unsigned)(ga & 0xffffffffu),
             (unsigned)((ga >> 32) & 0x1ffffffu) | (2u << 30) };
  v8i g1 = { (int)((1u << 16) | (1u << 19)),   // data_size=2B, iterate_enable
             (int)(256u << 16),                // tensor_dim0 = 256
             (int)(1u << 16),                  // tensor_dim1 = 1
             (int)(256u << 16),                // tile_dim0 = 256
             1,                                // tile_dim1 = 1
             256, 0, 0 };                      // tensor_dim0_stride
  v4i g2 = { 0,                                // tensor_dim2
             HROW,                             // lds_addr_increment (elements)
             256,                              // global_addr_increment (elements)
             (int)(15u << 16) };               // iterate_count = 15 -> 16 iters
  v4i g3 = (v4i)0;
  tdm_issue(g0, g1, g2, g3, store);
}

static __device__ __forceinline__ float fast_tanh(float x) {
#if __has_builtin(__builtin_amdgcn_tanhf)
  return __builtin_amdgcn_tanhf(x);
#elif __has_builtin(__builtin_amdgcn_tanh_f32)
  return __builtin_amdgcn_tanh_f32(x);
#else
  return tanhf(x);
#endif
}

static __device__ __forceinline__ v8f wmma_bf16(v16bf a, v16bf b, v8f c) {
  return __builtin_amdgcn_wmma_f32_16x16x32_bf16(false, a, false, b,
                                                 (short)0, c, false, false);
}

static __device__ __forceinline__ unsigned pack_bf16(float lo, float hi) {
  return (unsigned)__builtin_bit_cast(unsigned short, (__bf16)lo) |
         ((unsigned)__builtin_bit_cast(unsigned short, (__bf16)hi) << 16);
}

// A-fragment from a *pre-converted* row-major bf16 [256x256] weight image.
// Lane holds row j; K elems d0..d0+7 then d0+16..d0+23 (16-bit A 16x32 layout).
// Pure b128 loads + bit_cast: no fp32 copies exist -> frags stay resident.
static __device__ __forceinline__ v16bf
load_a_frag_bf(const unsigned short* __restrict__ Wbf, int j, int d0) {
  const unsigned short* p = Wbf + j * DHID + d0;
  U32x8 raw;
  raw.a = *(const u32x4*)(p);        // K = d0 .. d0+7
  raw.b = *(const u32x4*)(p + 16);   // K = d0+16 .. d0+23
  return __builtin_bit_cast(v16bf, raw);
}

// B-fragment (32x16 bf16, K = 8h..8h+7 then 16+8h..16+8h+7 per lane) from a padded-row LDS image.
static __device__ __forceinline__ v16bf load_b_frag(const unsigned short* base,
                                                    int ln, int kf, int hi) {
  const unsigned short* q = base + ln * HROW + kf * 32 + hi * 8;
  U32x8 raw;
  raw.a = *(const u32x4*)(q);
  raw.b = *(const u32x4*)(q + 16);
  return __builtin_bit_cast(v16bf, raw);
}

// ---------------------------------------------------------------------------
// Kernel 0: fp32 -> bf16 conversion (embedding table and weight matrices).
// 8 elements per thread; all sizes are exact multiples of 2048.
// ---------------------------------------------------------------------------
__global__ void __launch_bounds__(256)
k_cvt(const float* __restrict__ src, unsigned short* __restrict__ dst) {
  const int i = (blockIdx.x * 256 + threadIdx.x) * 8;
  const float4* p = (const float4*)(src + i);
  float4 a = p[0], b = p[1];
  u32x4 o;
  o.x = pack_bf16(a.x, a.y);
  o.y = pack_bf16(a.z, a.w);
  o.z = pack_bf16(b.x, b.y);
  o.w = pack_bf16(b.z, b.w);
  *(u32x4*)(dst + i) = o;
}

// ---------------------------------------------------------------------------
// Kernel 1: TDM gather-mode embedding lookup (bf16) + input projection GEMM.
// xp tile [n][j] (dense 512B rows in HBM, 528B padded rows in LDS) per step.
// grid = (4096/16 batch blocks, 8 time chunks), 256 threads (8 waves).
// ---------------------------------------------------------------------------
__global__ void __launch_bounds__(256)
k_proj(const int* __restrict__ x, const unsigned short* __restrict__ emb_bf,
       const unsigned short* __restrict__ wih_bf, const float* __restrict__ b_ih,
       const float* __restrict__ b_hh, unsigned short* __restrict__ xp) {
  __shared__ alignas(16) unsigned short gbuf[2][NB * HROW];   // gathered rows
  __shared__ alignas(16) unsigned short stage[2][NB * HROW];  // outgoing tile

  const int tid  = threadIdx.x;
  const int lane = tid & 31;
  const int wave = tid >> 5;
  const int hi   = lane >> 4;
  const int ln   = lane & 15;
  const int bblk = blockIdx.x;
  const int t0   = blockIdx.y * (SEQLEN / 8);
  const int t1   = t0 + (SEQLEN / 8);

  // Persistent bf16 A fragments of W_ih (2 j-tiles per wave) + fused bias.
  v16bf A[2][8];
  v8f   cbias[2];
#pragma unroll
  for (int jt = 0; jt < 2; ++jt) {
    const int jtb = wave * 32 + jt * 16;
#pragma unroll
    for (int kf = 0; kf < 8; ++kf)
      A[jt][kf] = load_a_frag_bf(wih_bf, jtb + ln, kf * 32 + hi * 8);
#pragma unroll
    for (int m = 0; m < 8; ++m) {
      const int j = jtb + hi * 8 + m;
      cbias[jt][m] = b_ih[j] + b_hh[j];
    }
  }

  // TDM gather: 16 bf16 embedding rows -> LDS, 16B pad per 512B row.
  auto issue_gather = [&](int t, int buf) {
    int myidx = 0;
    if (lane < 16) myidx = x[(bblk * NB + lane) * SEQLEN + t];
    int pk[8];
#pragma unroll
    for (int i = 0; i < 8; ++i) {
      int lo = __builtin_amdgcn_readlane(myidx, 2 * i);
      int up = __builtin_amdgcn_readlane(myidx, 2 * i + 1);
      pk[i] = (lo & 0xffff) | (up << 16);
    }
    unsigned lds = (unsigned)(size_t)&gbuf[buf][0];
    unsigned long long ga = (unsigned long long)(size_t)emb_bf;
    v4u g0 = { 0x80000001u,                       // count=1, gather_mode, 16b idx
               lds,
               (unsigned)(ga & 0xffffffffu),
               (unsigned)((ga >> 32) & 0x1ffffffu) | (2u << 30) };
    v8i g1 = { (int)((1u << 16) | (1u << 20) | (6u << 22) | (3u << 25)),
               (int)(256u << 16),                 // tensor_dim0 = 256
               (int)((unsigned)VOCAB << 16),      // tensor_dim1 = VOCAB
               (int)(256u << 16),                 // tile_dim0 = 256
               16,                                // tile_dim1 = #indices
               256,                               // tensor_dim0_stride
               0, 0 };
    v4i g2 = { pk[0], pk[1], pk[2], pk[3] };
    v4i g3 = { pk[4], pk[5], pk[6], pk[7] };
    tdm_issue(g0, g1, g2, g3, false);
  };

  if (wave == 0) issue_gather(t0, 0);

  for (int t = t0; t < t1; ++t) {
    if (wave == 0) {
      if (t == t0) __builtin_amdgcn_s_wait_tensorcnt((short)0);  // gather(t) done
      else         __builtin_amdgcn_s_wait_tensorcnt((short)1);  // store(t-1) may fly
    }
    __syncthreads();
    if (wave == 0 && t + 1 < t1) issue_gather(t + 1, (t + 1) & 1);

    const unsigned short* gb = &gbuf[t & 1][0];
    v8f acc0 = cbias[0], acc1 = cbias[1];
#pragma unroll
    for (int kf = 0; kf < 8; ++kf) {
      v16bf b = load_b_frag(gb, ln, kf, hi);
      acc0 = wmma_bf16(A[0][kf], b, acc0);
      acc1 = wmma_bf16(A[1][kf], b, acc1);
    }

    unsigned short* st = &stage[t & 1][0];
#pragma unroll
    for (int m = 0; m < 8; m += 2) {
      *(unsigned*)&st[ln * HROW + wave * 32      + hi * 8 + m] =
          pack_bf16(acc0[m], acc0[m + 1]);
      *(unsigned*)&st[ln * HROW + wave * 32 + 16 + hi * 8 + m] =
          pack_bf16(acc1[m], acc1[m + 1]);
    }
    __syncthreads();
    if (wave == 0)
      tdm_rows16((unsigned)(size_t)&stage[t & 1][0],
                 (unsigned long long)(size_t)xp +
                     (unsigned long long)(bblk * SEQLEN + t) * TILE_BYTES,
                 true);
  }
  if (wave == 0) __builtin_amdgcn_s_wait_tensorcnt((short)0);
}

// ---------------------------------------------------------------------------
// Kernel 2: time-serial recurrence h = tanh(xp_t + h W_hh^T), transposed
// (h^T_new = W_hh h^T); W_hh bf16 A-frags persist in VGPRs, h ping-pongs
// through LDS, x_proj tiles prefetched one step ahead via TDM iterate loads.
// grid = 256 blocks, 256 threads (8 waves).
// ---------------------------------------------------------------------------
__global__ void __launch_bounds__(256)
k_rnn(const unsigned short* __restrict__ xp,
      const unsigned short* __restrict__ whh_bf,
      const float* __restrict__ W_cls, const float* __restrict__ b_cls,
      float* __restrict__ out) {
  __shared__ alignas(16) unsigned short hbuf[2][NB * HROW];  // h^T bf16 [n][k]
  __shared__ alignas(16) unsigned short xpb[2][NB * HROW];   // x_proj [n][j]
  __shared__ float outl[NB * 5];                             // classifier acc

  const int tid  = threadIdx.x;
  const int lane = tid & 31;
  const int wave = tid >> 5;
  const int hi   = lane >> 4;
  const int ln   = lane & 15;
  const int bblk = blockIdx.x;

  // Persistent bf16 A fragments of W_hh.
  v16bf A[2][8];
#pragma unroll
  for (int jt = 0; jt < 2; ++jt) {
    const int jtb = wave * 32 + jt * 16;
#pragma unroll
    for (int kf = 0; kf < 8; ++kf)
      A[jt][kf] = load_a_frag_bf(whh_bf, jtb + ln, kf * 32 + hi * 8);
  }

  for (int i = tid; i < NB * HROW; i += 256) hbuf[0][i] = 0;   // h0 = 0
  if (tid < NB * 5) outl[tid] = b_cls[tid % 5];

  auto issue_load = [&](int t, int buf) {
    tdm_rows16((unsigned)(size_t)&xpb[buf][0],
               (unsigned long long)(size_t)xp +
                   (unsigned long long)(bblk * SEQLEN + t) * TILE_BYTES,
               false);
  };

  if (wave == 0) issue_load(0, 0);

  float hc0[8], hc1[8];
  for (int t = 0; t < SEQLEN; ++t) {
    if (wave == 0) __builtin_amdgcn_s_wait_tensorcnt((short)0);  // xp(t) ready
    __syncthreads();
    if (wave == 0 && t + 1 < SEQLEN) issue_load(t + 1, (t + 1) & 1);

    const unsigned short* hs = &hbuf[t & 1][0];
    unsigned short*       hd = &hbuf[(t + 1) & 1][0];
    const unsigned short* xt = &xpb[t & 1][0];

    // C init: one b128 per j-tile from the padded x_proj rows.
    v8bf xv0 = __builtin_bit_cast(
        v8bf, *(const u32x4*)(xt + ln * HROW + wave * 32 + hi * 8));
    v8bf xv1 = __builtin_bit_cast(
        v8bf, *(const u32x4*)(xt + ln * HROW + wave * 32 + 16 + hi * 8));
    v8f acc0, acc1;
#pragma unroll
    for (int m = 0; m < 8; ++m) {
      acc0[m] = (float)xv0[m];
      acc1[m] = (float)xv1[m];
    }
#pragma unroll
    for (int kf = 0; kf < 8; ++kf) {
      v16bf b = load_b_frag(hs, ln, kf, hi);
      acc0 = wmma_bf16(A[0][kf], b, acc0);
      acc1 = wmma_bf16(A[1][kf], b, acc1);
    }
#pragma unroll
    for (int m = 0; m < 8; ++m) {
      hc0[m] = fast_tanh(acc0[m]);
      hc1[m] = fast_tanh(acc1[m]);
    }
    // Write new h^T (bf16 pairs) for the next step's B operand.
#pragma unroll
    for (int m = 0; m < 8; m += 2) {
      *(unsigned*)&hd[ln * HROW + wave * 32      + hi * 8 + m] =
          pack_bf16(hc0[m], hc0[m + 1]);
      *(unsigned*)&hd[ln * HROW + wave * 32 + 16 + hi * 8 + m] =
          pack_bf16(hc1[m], hc1[m + 1]);
    }
  }

  // Classifier: logits[n, c] = sum_j h[n,j] * W_cls[c,j] + b_cls[c]
  __syncthreads();
#pragma unroll
  for (int c = 0; c < 5; ++c) {
    float p = 0.f;
#pragma unroll
    for (int m = 0; m < 8; ++m) {
      p += hc0[m] * W_cls[c * DHID + wave * 32      + hi * 8 + m];
      p += hc1[m] * W_cls[c * DHID + wave * 32 + 16 + hi * 8 + m];
    }
    atomicAdd(&outl[ln * 5 + c], p);
  }
  __syncthreads();
  if (tid < NB * 5)
    out[(bblk * NB + tid / 5) * 5 + (tid % 5)] = outl[tid];
}

// ---------------------------------------------------------------------------
extern "C" void kernel_launch(void* const* d_in, const int* in_sizes, int n_in,
                              void* d_out, int out_size, void* d_ws, size_t ws_size,
                              hipStream_t stream) {
  (void)in_sizes; (void)n_in; (void)out_size; (void)ws_size;
  const int*   x     = (const int*)d_in[0];
  const float* emb   = (const float*)d_in[1];
  const float* W_ih  = (const float*)d_in[2];
  const float* W_hh  = (const float*)d_in[3];
  const float* b_ih  = (const float*)d_in[4];
  const float* b_hh  = (const float*)d_in[5];
  const float* W_cls = (const float*)d_in[6];
  const float* b_cls = (const float*)d_in[7];
  float* out = (float*)d_out;

  unsigned short* xp     = (unsigned short*)d_ws;                // 256 MiB x_proj
  unsigned short* emb_bf = (unsigned short*)((char*)d_ws + XP_BYTES);
  unsigned short* wih_bf = emb_bf + (size_t)VOCAB * DHID;
  unsigned short* whh_bf = wih_bf + (size_t)DHID * DHID;

  k_cvt<<<(VOCAB * DHID) / 2048, 256, 0, stream>>>(emb, emb_bf);
  k_cvt<<<(DHID * DHID) / 2048, 256, 0, stream>>>(W_ih, wih_bf);
  k_cvt<<<(DHID * DHID) / 2048, 256, 0, stream>>>(W_hh, whh_bf);

  dim3 g1(4096 / NB, 8);          // batch blocks x time chunks
  k_proj<<<g1, 256, 0, stream>>>(x, emb_bf, wih_bf, b_ih, b_hh, xp);
  k_rnn<<<4096 / NB, 256, 0, stream>>>(xp, whh_bf, W_cls, b_cls, out);
}
